// VectorQuantizer_30743375905293
// MI455X (gfx1250) — matrix operand order repaired and checked by reading, not compiled, not run on hardware
//
#include <hip/hip_runtime.h>

typedef __attribute__((ext_vector_type(8)))  float  v8f;
typedef __attribute__((ext_vector_type(8)))  __bf16 v8bf;
typedef __attribute__((ext_vector_type(16))) __bf16 v16bf;

#define CAT16(a,b) __builtin_shufflevector(a,b,0,1,2,3,4,5,6,7,8,9,10,11,12,13,14,15)

#define NROWS 32768      // 32*1024
#define DIM   256
#define KCB   8192
#define Q_OFF 1
#define P_OFF 8388609    // 1 + 32768*256
#define E_OFF 8388610

// ---------------- zero fill of the 1 GiB one-hot encodings (8B-aligned base) ---------
__global__ __launch_bounds__(256) void vq_fill_enc_zero(float2* __restrict__ p, unsigned long long n2) {
    unsigned long long i = (unsigned long long)blockIdx.x * 2048ull + threadIdx.x;
    float2 z; z.x = 0.f; z.y = 0.f;
#pragma unroll
    for (int k = 0; k < 8; ++k) { if (i < n2) p[i] = z; i += 256; }
}

__global__ __launch_bounds__(256) void vq_zero_small(float* __restrict__ rowsq, int* __restrict__ hist) {
    int i = blockIdx.x * 256 + threadIdx.x;          // 160*256 = 40960 threads
    if (i < NROWS) rowsq[i] = 0.f;
    else if (i < NROWS + KCB) hist[i - NROWS] = 0;
}

// ---------------- f32 -> bf16 hi/lo split, row-major ---------------------------------
__device__ __forceinline__ void bf_split(float f, __bf16& h, __bf16& l) {
    h = (__bf16)f;
    l = (__bf16)(f - (float)h);
}

__global__ __launch_bounds__(256) void vq_cvt_inputs(const float* __restrict__ x,
                                                     __bf16* __restrict__ xhi, __bf16* __restrict__ xlo) {
    unsigned long long base = ((unsigned long long)blockIdx.x * 256ull + threadIdx.x) * 8ull;
    float4 a = *(const float4*)(x + base);
    float4 b = *(const float4*)(x + base + 4);
    float v[8] = {a.x,a.y,a.z,a.w,b.x,b.y,b.z,b.w};
    v8bf h, l;
#pragma unroll
    for (int k = 0; k < 8; ++k) { __bf16 hh, ll; bf_split(v[k], hh, ll); h[k] = hh; l[k] = ll; }
    *(v8bf*)(xhi + base) = h;
    *(v8bf*)(xlo + base) = l;
}

__global__ __launch_bounds__(256) void vq_cvt_codebook(const float* __restrict__ cb,
                                                       __bf16* __restrict__ chi, __bf16* __restrict__ clo,
                                                       float* __restrict__ nhalf) {
    int lane = threadIdx.x & 31, wave = threadIdx.x >> 5;
    int row  = blockIdx.x * 8 + wave;                // 1024 blocks * 8 waves = 8192 rows
    unsigned long long base = (unsigned long long)row * DIM + lane * 8;
    float4 a = *(const float4*)(cb + base);
    float4 b = *(const float4*)(cb + base + 4);
    float v[8] = {a.x,a.y,a.z,a.w,b.x,b.y,b.z,b.w};
    v8bf h, l; float s = 0.f;
#pragma unroll
    for (int k = 0; k < 8; ++k) { __bf16 hh, ll; bf_split(v[k], hh, ll); h[k] = hh; l[k] = ll; s += v[k]*v[k]; }
    *(v8bf*)(chi + base) = h;
    *(v8bf*)(clo + base) = l;
#pragma unroll
    for (int m = 16; m >= 1; m >>= 1) s += __shfl_xor(s, m, 32);
    if (lane == 0) nhalf[row] = 0.5f * s;
}

// ---------------- fused WMMA GEMM + argmin -------------------------------------------
// Each wave owns 16 rows of x (A, register-resident bf16 hi/lo for all D=256) and sweeps
// all 8192 codebook columns in 16-wide tiles, tracking running min of (0.5|e|^2 - x.e).
__global__ __launch_bounds__(256) void vq_argmin_wmma(const __bf16* __restrict__ xhi,
                                                      const __bf16* __restrict__ xlo,
                                                      const __bf16* __restrict__ chi,
                                                      const __bf16* __restrict__ clo,
                                                      const float*  __restrict__ nhalf,
                                                      int* __restrict__ idxOut) {
    const int lane  = threadIdx.x & 31;
    const int wave  = threadIdx.x >> 5;
    const int lmod  = lane & 15;
    const int lhalf = lane >> 4;
    const int mBase = blockIdx.x * 128 + wave * 16;
    const int row   = mBase + lmod;

    // A fragments (16-bit A 16x32, contiguous K): lanes 0-15 hold K0..15, lanes 16-31 K16..31
    v16bf Ahi[8], Alo[8];
#pragma unroll
    for (int c = 0; c < 8; ++c) {
        const __bf16* ph = xhi + (unsigned long long)row * DIM + c * 32 + lhalf * 16;
        const __bf16* pl = xlo + (unsigned long long)row * DIM + c * 32 + lhalf * 16;
        v8bf h0 = *(const v8bf*)ph, h1 = *(const v8bf*)(ph + 8);
        v8bf l0 = *(const v8bf*)pl, l1 = *(const v8bf*)(pl + 8);
        Ahi[c] = CAT16(h0, h1);
        Alo[c] = CAT16(l0, l1);
    }

    float minv[8]; int mini[8];
#pragma unroll
    for (int r = 0; r < 8; ++r) { minv[r] = 3.4e38f; mini[r] = 0; }

    for (int t = 0; t < KCB / 16; ++t) {
        const int n = t * 16 + lmod;               // this lane's codebook column
        const __bf16* bh = chi + (unsigned long long)n * DIM + lhalf * 16;
        const __bf16* bl = clo + (unsigned long long)n * DIM + lhalf * 16;
        __builtin_prefetch(bh + 16 * DIM, 0, 1);   // next tile, same lane slot
        __builtin_prefetch(bl + 16 * DIM, 0, 1);
        float nh = nhalf[n];
        v8f acc0 = {0.f,0.f,0.f,0.f,0.f,0.f,0.f,0.f};
        v8f acc1 = acc0, acc2 = acc0;
#pragma unroll
        for (int c = 0; c < 8; ++c) {
            // B fragments (32x16 bf16): lanes 0-15 K0..15, lanes 16-31 K16..31, N = lane%16
            v8bf h0 = *(const v8bf*)(bh + c * 32);
            v8bf h1 = *(const v8bf*)(bh + c * 32 + 8);
            v8bf l0 = *(const v8bf*)(bl + c * 32);
            v8bf l1 = *(const v8bf*)(bl + c * 32 + 8);
            v16bf Bh = CAT16(h0, h1);
            v16bf Bl = CAT16(l0, l1);
            // x.e ~= hi*hi + hi*lo + lo*hi  (three independent accumulator chains)
            acc0 = __builtin_amdgcn_wmma_f32_16x16x32_bf16(false, Ahi[c], false, Bh, (short)0, acc0, false, false);
            acc1 = __builtin_amdgcn_wmma_f32_16x16x32_bf16(false, Ahi[c], false, Bl, (short)0, acc1, false, false);
            acc2 = __builtin_amdgcn_wmma_f32_16x16x32_bf16(false, Alo[c], false, Bh, (short)0, acc2, false, false);
        }
#pragma unroll
        for (int r = 0; r < 8; ++r) {              // C layout: VGPR r -> row r+8*lhalf, col lmod
            float s = nh - (acc0[r] + acc1[r] + acc2[r]);
            if (s < minv[r]) { minv[r] = s; mini[r] = n; }
        }
    }

    // reduce argmin across the 16 column-lanes of each half (lowest index on ties)
#pragma unroll
    for (int r = 0; r < 8; ++r) {
        float v = minv[r]; int i = mini[r];
#pragma unroll
        for (int m = 1; m < 16; m <<= 1) {
            float ov = __shfl_xor(v, m, 32);
            int   oi = __shfl_xor(i, m, 32);
            if (ov < v || (ov == v && oi < i)) { v = ov; i = oi; }
        }
        if (lmod == 0) idxOut[mBase + r + 8 * lhalf] = i;
    }
}

// ---------------- gather, straight-through output, one-hot, hist, per-row MSE --------
__global__ __launch_bounds__(256) void vq_quantize(const float* __restrict__ x,
                                                   const float* __restrict__ cb,
                                                   const int* __restrict__ idxArr,
                                                   int* __restrict__ hist,
                                                   float* __restrict__ rowsq,
                                                   float* __restrict__ out) {
    int lane = threadIdx.x & 31, wave = threadIdx.x >> 5;
    int wgl  = blockIdx.x * 8 + wave;                 // 128 blocks * 8 waves = 1024 waves
    float* qst = out + Q_OFF;
    float* enc = out + E_OFF;
    for (int j = 0; j < 32; ++j) {
        int n  = wgl * 32 + j;
        int id = idxArr[n];
        const float* xr = x  + (unsigned long long)n  * DIM + lane * 8;
        const float* er = cb + (unsigned long long)id * DIM + lane * 8;
        float4 x0 = *(const float4*)xr, x1 = *(const float4*)(xr + 4);
        float4 e0 = *(const float4*)er, e1 = *(const float4*)(er + 4);
        float xv[8] = {x0.x,x0.y,x0.z,x0.w,x1.x,x1.y,x1.z,x1.w};
        float ev[8] = {e0.x,e0.y,e0.z,e0.w,e1.x,e1.y,e1.z,e1.w};
        float* o = qst + (unsigned long long)n * DIM + lane * 8;   // base offset 1 -> scalar stores
        float s = 0.f;
#pragma unroll
        for (int k = 0; k < 8; ++k) {
            float d = ev[k] - xv[k];
            s += d * d;
            o[k] = xv[k] + d;                         // x + sg(q - x), as the reference computes it
        }
#pragma unroll
        for (int m = 16; m >= 1; m >>= 1) s += __shfl_xor(s, m, 32);
        if (lane == 0) {
            rowsq[n] = s;                             // deterministic per-row partial
            atomicAdd(&hist[id], 1);                  // integer atomic: deterministic value
            enc[(unsigned long long)n * KCB + id] = 1.0f;
        }
    }
}

// ---------------- deterministic finalize: vq_loss and perplexity ---------------------
__global__ __launch_bounds__(1024) void vq_finalize(const float* __restrict__ rowsq,
                                                    const int* __restrict__ hist,
                                                    float* __restrict__ out) {
    __shared__ float red[1024];
    int t = threadIdx.x;
    float s = 0.f;
    for (int j = 0; j < NROWS / 1024; ++j) s += rowsq[t + j * 1024];
    red[t] = s; __syncthreads();
    for (int w = 512; w > 0; w >>= 1) { if (t < w) red[t] += red[t + w]; __syncthreads(); }
    if (t == 0) out[0] = 1.25f * red[0] / ((float)NROWS * (float)DIM);  // q_loss + 0.25*e_loss
    __syncthreads();
    float e = 0.f;
    for (int j = 0; j < KCB / 1024; ++j) {
        float p = (float)hist[t + j * 1024] * (1.0f / (float)NROWS);
        e += p * logf(p + 1e-10f);
    }
    red[t] = e; __syncthreads();
    for (int w = 512; w > 0; w >>= 1) { if (t < w) red[t] += red[t + w]; __syncthreads(); }
    if (t == 0) out[P_OFF] = expf(-red[0]);
}

extern "C" void kernel_launch(void* const* d_in, const int* in_sizes, int n_in,
                              void* d_out, int out_size, void* d_ws, size_t ws_size,
                              hipStream_t stream) {
    const float* x  = (const float*)d_in[0];   // [32,1024,256] f32
    const float* cb = (const float*)d_in[1];   // [8192,256] f32
    float* out = (float*)d_out;

    char* w = (char*)d_ws;
    __bf16* xhi  = (__bf16*)(w);                                   // 16 MB
    __bf16* xlo  = (__bf16*)(w + (16ull << 20));                   // 16 MB
    __bf16* chi  = (__bf16*)(w + (32ull << 20));                   //  4 MB
    __bf16* clo  = (__bf16*)(w + (36ull << 20));                   //  4 MB
    float*  nhalf= (float*) (w + (40ull << 20));                   // 32 KB
    int*    idxA = (int*)   (w + (40ull << 20) +  32ull * 1024);   // 128 KB
    int*    hist = (int*)   (w + (40ull << 20) + 160ull * 1024);   // 32 KB
    float*  rowsq= (float*) (w + (40ull << 20) + 192ull * 1024);   // 128 KB

    // 1) zero the 1 GiB one-hot block (8-byte aligned base at out+E_OFF)
    vq_fill_enc_zero<<<65536, 256, 0, stream>>>((float2*)(out + E_OFF),
                                                (unsigned long long)NROWS * KCB / 2ull);
    // 2) zero per-row loss partials + histogram
    vq_zero_small<<<160, 256, 0, stream>>>(rowsq, hist);
    // 3) bf16 hi/lo pre-split of inputs and codebook (+ 0.5*|e|^2)
    vq_cvt_inputs<<<4096, 256, 0, stream>>>(x, xhi, xlo);
    vq_cvt_codebook<<<1024, 256, 0, stream>>>(cb, chi, clo, nhalf);
    // 4) fused WMMA distance GEMM + argmin
    vq_argmin_wmma<<<256, 256, 0, stream>>>(xhi, xlo, chi, clo, nhalf, idxA);
    // 5) gather + straight-through output + one-hot scatter + hist + row MSE
    vq_quantize<<<128, 256, 0, stream>>>(x, cb, idxA, hist, rowsq, out);
    // 6) scalars: vq_loss, perplexity
    vq_finalize<<<1, 1024, 0, stream>>>(rowsq, hist, out);
}
